// CausalSelfAttention_75763223101714
// MI455X (gfx1250) — matrix operand order repaired and checked by reading, not compiled, not run on hardware
//
#include <hip/hip_runtime.h>

// ---------------------------------------------------------------------------
// CDNA5 (gfx1250) causal self-attention, bf16 WMMA path, f32 accumulate.
// GEMMs use async global->LDS staging (ASYNCcnt) with double buffering.
// ---------------------------------------------------------------------------

typedef __attribute__((ext_vector_type(16))) __bf16 v16bf;
typedef __attribute__((ext_vector_type(8)))  float  v8f;

#define DEV __device__ __forceinline__

static constexpr int SEQ    = 2048;
static constexpr int DMODEL = 2048;
static constexpr int NH     = 16;
static constexpr int DH     = 128;
static constexpr int BATCH  = 2;
static constexpr int MTOK   = BATCH * SEQ;   // 4096 rows for projections
static constexpr int LDSP   = 40;           // padded LDS row (32 data + 8 pad)

// float -> bf16 bits, round-to-nearest-even
DEV unsigned short f2bf(float f) {
  unsigned u = __builtin_bit_cast(unsigned, f);
  u += 0x7FFFu + ((u >> 16) & 1u);
  return (unsigned short)(u >> 16);
}

union FragU { uint4 q[2]; v16bf v; };

// A fragment (16x32 bf16): lane = row (lane&15); 16 contiguous K values
// starting at 2*koff = 16*(lane>>4)  (two 16B loads).
DEV v16bf ldfragA(const unsigned short* rowk0, int koff) {
  FragU f;
  f.q[0] = *(const uint4*)(rowk0 + 2 * koff);
  f.q[1] = *(const uint4*)(rowk0 + 2 * koff + 8);
  return f.v;
}
// B fragment (32x16 bf16) supplied as rows of B^T: lane = column (lane&15);
// 16 contiguous K values starting at 16*(lane>>4)  (one 32B region).
DEV v16bf ldfragB(const unsigned short* p) {
  FragU f;
  f.q[0] = *(const uint4*)(p);
  f.q[1] = *(const uint4*)(p + 8);
  return f.v;
}

DEV v8f wmma_bf16(v16bf a, v16bf b, v8f c) {
  return __builtin_amdgcn_wmma_f32_16x16x32_bf16(
      /*neg_a=*/false, a, /*neg_b=*/false, b,
      /*c_mod=*/(short)0, c, /*reuse_a=*/false, /*reuse_b=*/false);
}

// Async copy of one 16B chunk: global -> LDS (ASYNCcnt tracked).
DEV void async_cp16(unsigned ldsOff, const unsigned short* g) {
  asm volatile("global_load_async_to_lds_b128 %0, %1, off"
               :: "v"(ldsOff), "v"(g) : "memory");
}

// ---------------------------------------------------------------------------
// f32 -> bf16 elementwise convert
// ---------------------------------------------------------------------------
__global__ __launch_bounds__(256) void cvt_f32_bf16(
    const float* __restrict__ in, unsigned short* __restrict__ out, size_t n) {
  size_t i = (size_t)blockIdx.x * blockDim.x + threadIdx.x;
  size_t stride = (size_t)gridDim.x * blockDim.x;
  for (; i < n; i += stride) out[i] = f2bf(in[i]);
}

// ---------------------------------------------------------------------------
// W (K x N f32, row-major) -> Wt (N x K bf16, row-major)
// ---------------------------------------------------------------------------
__global__ __launch_bounds__(256) void transpose_cvt(
    const float* __restrict__ W, unsigned short* __restrict__ Wt,
    int K, int N) {
  __shared__ float tile[32][33];
  const int k0 = blockIdx.y * 32, n0 = blockIdx.x * 32;
  const int tx = threadIdx.x & 31, ty = threadIdx.x >> 5;  // ty: 0..7
  #pragma unroll
  for (int i = ty; i < 32; i += 8)
    tile[i][tx] = W[(size_t)(k0 + i) * N + (n0 + tx)];
  __syncthreads();
  #pragma unroll
  for (int i = ty; i < 32; i += 8)
    Wt[(size_t)(n0 + i) * K + (k0 + tx)] = f2bf(tile[tx][i]);
}

// ---------------------------------------------------------------------------
// GEMM:  C[M,N] = A[M,K](bf16) * Bt[N,K](bf16)^T + bias
// 256 threads = 8 waves (4 M-rows x 2 N-cols); block tile 128x128;
// wave tile 32x64; K-step 32.  A/B tiles staged into LDS with
// global_load_async_to_lds_b128, double buffered.
// EPI 1: bf16 -> [b][h][s][d]   (Q, K)
// EPI 2: bf16 -> [b][h][d][s]   (V transposed)
// EPI 3: f32  -> [m][n]         (output projection)
// EPI 0: bf16 -> [m][n]
// ---------------------------------------------------------------------------
template <int EPI>
__global__ __launch_bounds__(256) void gemm_wmma_bf16(
    const unsigned short* __restrict__ A,
    const unsigned short* __restrict__ Bt,
    const float* __restrict__ bias,
    void* __restrict__ out,
    int M, int N, int K) {
  __shared__ __attribute__((aligned(16))) unsigned short Abuf[2][128][LDSP];
  __shared__ __attribute__((aligned(16))) unsigned short Bbuf[2][128][LDSP];

  const int t    = threadIdx.x;
  const int lane = t & 31;
  const int wave = t >> 5;
  const int hi = lane >> 4;   // lane half
  const int lr = lane & 15;
  const int wm = wave & 3;    // wave M-row (0..3)
  const int wn = wave >> 2;   // wave N-col (0..1)
  const int mBase = blockIdx.y * 128;
  const int nBase = blockIdx.x * 128;
  const int koff = hi * 8;

  // stage K-step tile (128 rows x 32 cols) into LDS buffer `b`
  auto stage = [&](int b, int k0) {
    #pragma unroll
    for (int i = 0; i < 2; ++i) {
      const int c = t + i * 256;          // chunk id 0..511
      const int row = c >> 2;
      const int o = (c & 3) * 8;          // element offset within row
      async_cp16((unsigned)(uintptr_t)&Abuf[b][row][o],
                 A + (size_t)(mBase + row) * K + k0 + o);
      async_cp16((unsigned)(uintptr_t)&Bbuf[b][row][o],
                 Bt + (size_t)(nBase + row) * K + k0 + o);
    }
  };

  v8f acc[8] = {};

  stage(0, 0);
  int buf = 0;
  for (int k0 = 0; k0 < K; k0 += 32) {
    const bool hasNext = (k0 + 32) < K;
    if (hasNext) {
      stage(buf ^ 1, k0 + 32);
      asm volatile("s_wait_asynccnt 0x4" ::: "memory");
    } else {
      asm volatile("s_wait_asynccnt 0x0" ::: "memory");
    }
    __syncthreads();   // whole tile visible to all waves

    v16bf a0 = ldfragA(&Abuf[buf][wm * 32 + lr][0], koff);
    v16bf a1 = ldfragA(&Abuf[buf][wm * 32 + 16 + lr][0], koff);
    #pragma unroll
    for (int nt = 0; nt < 4; ++nt) {
      v16bf bfr = ldfragB(&Bbuf[buf][wn * 64 + nt * 16 + lr][hi * 16]);
      acc[nt]     = wmma_bf16(a0, bfr, acc[nt]);
      acc[4 + nt] = wmma_bf16(a1, bfr, acc[4 + nt]);
    }

    asm volatile("s_wait_dscnt 0x0" ::: "memory");
    __syncthreads();   // safe to overwrite this buffer next round
    buf ^= 1;
  }

  #pragma unroll
  for (int s = 0; s < 2; ++s) {
    #pragma unroll
    for (int nt = 0; nt < 4; ++nt) {
      #pragma unroll
      for (int r = 0; r < 8; ++r) {
        const int m = mBase + wm * 32 + s * 16 + r + 8 * hi;
        const int n = nBase + wn * 64 + nt * 16 + lr;
        const float v = acc[s * 4 + nt][r] + bias[n];
        if (EPI == 0) {
          ((unsigned short*)out)[(size_t)m * N + n] = f2bf(v);
        } else if (EPI == 1) {            // Q,K: [b][h][s][d]
          const int b = m >> 11, sTok = m & (SEQ - 1);
          const int h = n >> 7, d = n & (DH - 1);
          ((unsigned short*)out)[((((size_t)b * NH + h) * SEQ) + sTok) * DH + d] =
              f2bf(v);
        } else if (EPI == 2) {            // V^T: [b][h][d][s]
          const int b = m >> 11, sTok = m & (SEQ - 1);
          const int h = n >> 7, d = n & (DH - 1);
          ((unsigned short*)out)[(((size_t)b * NH + h) * DH + d) * SEQ + sTok] =
              f2bf(v);
        } else {                          // f32 output
          ((float*)out)[(size_t)m * N + n] = v;
        }
      }
    }
  }
}

// ---------------------------------------------------------------------------
// Flash attention: one wave = 16 query rows; block = 8 waves = 128 queries.
// grid = (SEQ/128, B*H). Online softmax over 32-key blocks; causal masked.
// ---------------------------------------------------------------------------
__global__ __launch_bounds__(256) void flash_attn_wmma(
    const unsigned short* __restrict__ Q,   // [B*H][S][DH]
    const unsigned short* __restrict__ Kd,  // [B*H][S][DH]
    const unsigned short* __restrict__ Vt,  // [B*H][DH][S]
    unsigned short* __restrict__ O) {       // [B*S][DMODEL]
  __shared__ __attribute__((aligned(16))) unsigned short pTile[8][16][32];

  const int lane = threadIdx.x & 31;
  const int wave = threadIdx.x >> 5;
  const int hi = lane >> 4;
  const int lr = lane & 15;
  const int bh = blockIdx.y;            // 0..31
  const int qBase = blockIdx.x * 128 + wave * 16;
  const float scale = 0.08838834764831845f;  // 1/sqrt(128)

  const unsigned short* qHead = Q  + (size_t)bh * SEQ * DH;
  const unsigned short* kHead = Kd + (size_t)bh * SEQ * DH;
  const unsigned short* vHead = Vt + (size_t)bh * DH * SEQ;

  // Preload 16x128 Q tile as 4 A-fragments (K = 0..127)
  v16bf qf[4];
  {
    const unsigned short* qRow = qHead + (size_t)(qBase + lr) * DH;
    #pragma unroll
    for (int i = 0; i < 4; ++i) qf[i] = ldfragA(qRow + 32 * i, hi * 8);
  }

  v8f o[8] = {};                // 16 x 128 output accumulator (8 N-tiles)
  float mrow[8], lrow[8];
  #pragma unroll
  for (int r = 0; r < 8; ++r) { mrow[r] = -3.0e38f; lrow[r] = 0.0f; }

  const int kvEnd = qBase + 16;           // keys strictly < kvEnd can be valid
  for (int kb = 0; kb < kvEnd; kb += 32) {
    // ---- S = Q * K^T : two 16x16 score tiles (keys kb.., kb+16..) ----
    v8f s0 = {}, s1 = {};
    #pragma unroll
    for (int i = 0; i < 4; ++i) {
      v16bf b0 = ldfragB(kHead + (size_t)(kb + lr) * DH + 32 * i + hi * 16);
      s0 = wmma_bf16(qf[i], b0, s0);
      v16bf b1 = ldfragB(kHead + (size_t)(kb + 16 + lr) * DH + 32 * i + hi * 16);
      s1 = wmma_bf16(qf[i], b1, s1);
    }

    // ---- online softmax (row stats per M; N striped over half-wave) ----
    #pragma unroll
    for (int r = 0; r < 8; ++r) {
      const int q = qBase + r + 8 * hi;
      float v0 = s0[r] * scale; if (kb + lr      > q) v0 = -1.0e30f;
      float v1 = s1[r] * scale; if (kb + 16 + lr > q) v1 = -1.0e30f;
      float mx = fmaxf(v0, v1);
      #pragma unroll
      for (int off = 1; off < 16; off <<= 1)
        mx = fmaxf(mx, __shfl_xor(mx, off, 32));
      const float mNew = fmaxf(mrow[r], mx);
      const float p0 = __expf(v0 - mNew);
      const float p1 = __expf(v1 - mNew);
      float rs = p0 + p1;
      #pragma unroll
      for (int off = 1; off < 16; off <<= 1) rs += __shfl_xor(rs, off, 32);
      const float alpha = __expf(mrow[r] - mNew);
      mrow[r] = mNew;
      lrow[r] = lrow[r] * alpha + rs;
      #pragma unroll
      for (int nt = 0; nt < 8; ++nt) o[nt][r] = o[nt][r] * alpha;
      // C-layout -> LDS (row-major 16x32 P tile, per wave)
      pTile[wave][r + 8 * hi][lr]      = f2bf(p0);
      pTile[wave][r + 8 * hi][16 + lr] = f2bf(p1);
    }
    asm volatile("s_wait_dscnt 0x0" ::: "memory");

    // ---- reload P as A-fragment, O += P * V ----
    v16bf pf = ldfragA(&pTile[wave][lr][0], hi * 8);
    asm volatile("" ::: "memory");
    #pragma unroll
    for (int nt = 0; nt < 8; ++nt) {
      v16bf vfr = ldfragB(vHead + (size_t)(nt * 16 + lr) * SEQ + kb + hi * 16);
      o[nt] = wmma_bf16(pf, vfr, o[nt]);
    }
    asm volatile("" ::: "memory");
  }

  // ---- finalize & store to [b][s][h*128+d] (bf16) ----
  const int b = bh >> 4, h = bh & 15;
  float inv[8];
  #pragma unroll
  for (int r = 0; r < 8; ++r) inv[r] = 1.0f / lrow[r];
  #pragma unroll
  for (int nt = 0; nt < 8; ++nt) {
    #pragma unroll
    for (int r = 0; r < 8; ++r) {
      const int s = qBase + r + 8 * hi;
      const int col = h * DH + nt * 16 + lr;
      O[((size_t)b * SEQ + s) * DMODEL + col] = f2bf(o[nt][r] * inv[r]);
    }
  }
}

// ---------------------------------------------------------------------------
// Host launcher
// ---------------------------------------------------------------------------
extern "C" void kernel_launch(void* const* d_in, const int* in_sizes, int n_in,
                              void* d_out, int out_size, void* d_ws,
                              size_t ws_size, hipStream_t stream) {
  (void)in_sizes; (void)n_in; (void)out_size; (void)ws_size;
  const float* x  = (const float*)d_in[0];
  // d_in[1] = mask (unused; causal mask applied analytically)
  const float* Wq = (const float*)d_in[2];
  const float* bq = (const float*)d_in[3];
  const float* Wk = (const float*)d_in[4];
  const float* bk = (const float*)d_in[5];
  const float* Wv = (const float*)d_in[6];
  const float* bv = (const float*)d_in[7];
  const float* Wo = (const float*)d_in[8];
  const float* bo = (const float*)d_in[9];
  float* out = (float*)d_out;

  const size_t xElems = (size_t)MTOK * DMODEL;      // 8M
  const size_t wElems = (size_t)DMODEL * DMODEL;    // 4M

  char* ws = (char*)d_ws;
  unsigned short* xb  = (unsigned short*)ws; ws += xElems * 2;  // reused as attnO
  unsigned short* WqT = (unsigned short*)ws; ws += wElems * 2;
  unsigned short* WkT = (unsigned short*)ws; ws += wElems * 2;
  unsigned short* WvT = (unsigned short*)ws; ws += wElems * 2;
  unsigned short* WoT = (unsigned short*)ws; ws += wElems * 2;
  unsigned short* Qb  = (unsigned short*)ws; ws += xElems * 2;
  unsigned short* Kb  = (unsigned short*)ws; ws += xElems * 2;
  unsigned short* Vtb = (unsigned short*)ws; ws += xElems * 2;

  cvt_f32_bf16<<<4096, 256, 0, stream>>>(x, xb, xElems);

  dim3 tGrid(DMODEL / 32, DMODEL / 32);
  transpose_cvt<<<tGrid, 256, 0, stream>>>(Wq, WqT, DMODEL, DMODEL);
  transpose_cvt<<<tGrid, 256, 0, stream>>>(Wk, WkT, DMODEL, DMODEL);
  transpose_cvt<<<tGrid, 256, 0, stream>>>(Wv, WvT, DMODEL, DMODEL);
  transpose_cvt<<<tGrid, 256, 0, stream>>>(Wo, WoT, DMODEL, DMODEL);

  dim3 gGrid(DMODEL / 128, MTOK / 128);   // (16, 32)
  gemm_wmma_bf16<1><<<gGrid, 256, 0, stream>>>(xb, WqT, bq, Qb,  MTOK, DMODEL, DMODEL);
  gemm_wmma_bf16<1><<<gGrid, 256, 0, stream>>>(xb, WkT, bk, Kb,  MTOK, DMODEL, DMODEL);
  gemm_wmma_bf16<2><<<gGrid, 256, 0, stream>>>(xb, WvT, bv, Vtb, MTOK, DMODEL, DMODEL);

  // flash attention writes bf16 attn output; reuse xb (QKV GEMMs done reading)
  unsigned short* attnO = xb;
  dim3 fGrid(SEQ / 128, BATCH * NH);     // (16, 32)
  flash_attn_wmma<<<fGrid, 256, 0, stream>>>(Qb, Kb, Vtb, attnO);

  gemm_wmma_bf16<3><<<gGrid, 256, 0, stream>>>(attnO, WoT, bo, out, MTOK, DMODEL, DMODEL);
}
